// KDMLP_17617955848556
// MI455X (gfx1250) — matrix operand order repaired and checked by reading, not compiled, not run on hardware
//
#include <hip/hip_runtime.h>

typedef __attribute__((ext_vector_type(16))) _Float16 v16h;
typedef __attribute__((ext_vector_type(8)))  _Float16 v8h;
typedef __attribute__((ext_vector_type(8)))  float    v8f;
typedef __attribute__((ext_vector_type(4)))  float    v4f;

#define B_TOTAL   65536
#define TILE_B    32
#define IN0       448
#define H1DIM     256
#define H2DIM     128
#define H3DIM     64
#define KNN       20

// padded LDS strides (in halves); each row start 16B aligned, banks conflict-free
#define XSTR   456
#define H1STR  264
#define H2STR  136
#define H3STR  72

// workspace layout (halves)
#define W1T_OFF  0
#define W1T_SZ   (H1DIM * IN0)      // 114688
#define W2T_OFF  (W1T_OFF + W1T_SZ)
#define W2T_SZ   (H2DIM * H1DIM)    // 32768
#define W3T_OFF  (W2T_OFF + W2T_SZ)
#define W3T_SZ   (H3DIM * H2DIM)    // 8192
#define WS_TOTAL (W3T_OFF + W3T_SZ) // 155648 halves

#define WMMA_F16(a, b, c) \
  __builtin_amdgcn_wmma_f32_16x16x32_f16(false, (a), false, (b), (short)0, (c), false, false)

// ---- fragment helpers (ISA 7.12.2 layouts, wave32) -------------------------

// A 16x32 f16: lane<16 -> row M=lane, K {0..7} then {16..23}; lane>=16 -> K {8..15},{24..31}
__device__ __forceinline__ v16h ld_a_lds(const _Float16* base, int stride, int mrow,
                                         int kc, int lane) {
  const int r  = mrow + (lane & 15);
  const int ks = kc * 32 + ((lane >> 4) << 3);
  const _Float16* p = base + r * stride + ks;
  v16h a;
  ((v8h*)&a)[0] = *(const v8h*)p;
  ((v8h*)&a)[1] = *(const v8h*)(p + 16);
  return a;
}

// B 32x16 f16 from N-major weights Wt[N][Kdim]: lane<16 -> col N, K {0..7},{16..23}; lane>=16 -> K {8..15},{24..31}
__device__ __forceinline__ v16h ld_b_glb(const _Float16* Wt, int Kdim, int n,
                                         int kc, int lane) {
  const int c  = n + (lane & 15);
  const int ks = kc * 32 + ((lane >> 4) << 3);
  const _Float16* p = Wt + (long)c * Kdim + ks;
  v16h b;
  ((v8h*)&b)[0] = *(const v8h*)p;
  ((v8h*)&b)[1] = *(const v8h*)(p + 16);
  return b;
}

__device__ __forceinline__ v8f bias_frag(const float* bias, int n, int lane) {
  const float bv = bias[n + (lane & 15)];
  v8f c;
#pragma unroll
  for (int v = 0; v < 8; ++v) c[v] = bv;
  return c;
}

// C/D frag: VGPR v -> M = v + 8*(lane>=16), N = lane&15
__device__ __forceinline__ void st_c_relu(_Float16* H, int stride, int mrow, int n,
                                          int lane, v8f c) {
  const int col = n + (lane & 15);
  const int mb  = mrow + ((lane >> 4) << 3);
#pragma unroll
  for (int v = 0; v < 8; ++v) {
    float x = c[v];
    x = x > 0.0f ? x : 0.0f;
    H[(mb + v) * stride + col] = (_Float16)x;
  }
}

// ---- weight prep: f32 [K,N] row-major -> f16 [N,K] --------------------------

__global__ void kd_prep_weights(const float* __restrict__ W1,
                                const float* __restrict__ W2,
                                const float* __restrict__ W3,
                                _Float16* __restrict__ ws) {
  int i = blockIdx.x * blockDim.x + threadIdx.x;
  if (i < W1T_SZ) {
    const int n = i / IN0, k = i % IN0;
    ws[W1T_OFF + i] = (_Float16)W1[k * H1DIM + n];
    return;
  }
  i -= W1T_SZ;
  if (i < W2T_SZ) {
    const int n = i / H1DIM, k = i % H1DIM;
    ws[W2T_OFF + i] = (_Float16)W2[k * H2DIM + n];
    return;
  }
  i -= W2T_SZ;
  if (i < W3T_SZ) {
    const int n = i / H2DIM, k = i % H2DIM;
    ws[W3T_OFF + i] = (_Float16)W3[k * H3DIM + n];
  }
}

// ---- masked mean pool: 8 threads per row, 8 cols/thread, one b128 LDS store -

__device__ __forceinline__ void pool_row(const float* __restrict__ ek,
                                         const int* __restrict__ idx, int vc,
                                         _Float16* __restrict__ dst, int t) {
  float acc[8];
#pragma unroll
  for (int j = 0; j < 8; ++j) acc[j] = 0.0f;
  for (int k = 0; k < vc; ++k) {
    const float* p = ek + (long)idx[k] * H3DIM + t * 8;
    v4f lo = *(const v4f*)p;
    v4f hi = *(const v4f*)(p + 4);
#pragma unroll
    for (int j = 0; j < 4; ++j) { acc[j] += lo[j]; acc[4 + j] += hi[j]; }
  }
  const float s = (vc > 0) ? (1.0f / (float)vc) : 0.0f;
  v8h h;
#pragma unroll
  for (int j = 0; j < 8; ++j) h[j] = (_Float16)(acc[j] * s);
  *(v8h*)(dst + t * 8) = h;     // 16B aligned: dst is row+multiple-of-64 halves
}

// convert 16 f32 (two v4f pairs) -> two packed v8h and store as 2x b128
__device__ __forceinline__ void cvt16_store(const float* __restrict__ src,
                                            _Float16* __restrict__ dst) {
  v4f a0 = *(const v4f*)(src + 0);
  v4f a1 = *(const v4f*)(src + 4);
  v4f a2 = *(const v4f*)(src + 8);
  v4f a3 = *(const v4f*)(src + 12);
  v8h lo, hi;
#pragma unroll
  for (int q = 0; q < 4; ++q) {
    lo[q] = (_Float16)a0[q]; lo[4 + q] = (_Float16)a1[q];
    hi[q] = (_Float16)a2[q]; hi[4 + q] = (_Float16)a3[q];
  }
  *(v8h*)(dst)     = lo;
  *(v8h*)(dst + 8) = hi;
}

// ---- fused kernel: gather+pool -> 3 WMMA GEMM layers -> dot ----------------

__global__ __launch_bounds__(256, 1)
void kd_fused(const int* __restrict__ user, const int* __restrict__ item,
              const int* __restrict__ pk, const int* __restrict__ tk,
              const int* __restrict__ ik,
              const int* __restrict__ vpk, const int* __restrict__ vtk,
              const int* __restrict__ vik,
              const float* __restrict__ emb_user,
              const float* __restrict__ emb_item,
              const float* __restrict__ emb_k,
              const _Float16* __restrict__ W1t, const float* __restrict__ b1,
              const _Float16* __restrict__ W2t, const float* __restrict__ b2,
              const _Float16* __restrict__ W3t, const float* __restrict__ b3,
              const float* __restrict__ Wp, const float* __restrict__ bp,
              float* __restrict__ out) {
  __shared__ __align__(16) _Float16 sX[TILE_B * XSTR];
  __shared__ __align__(16) _Float16 sH1[TILE_B * H1STR];
  __shared__ __align__(16) _Float16 sH2[TILE_B * H2STR];
  __shared__ __align__(16) _Float16 sH3[TILE_B * H3STR];

  const int tid  = threadIdx.x;
  const int lane = tid & 31;
  const int wv   = tid >> 5;           // 0..7

  // ---------------- phase 1: build X tile [32,448] in f16 LDS ---------------
  {
    const int row = tid >> 3;          // 0..31
    const int t   = tid & 7;           // 0..7
    const long r  = (long)blockIdx.x * TILE_B + row;
    _Float16* xr  = sX + row * XSTR;

    cvt16_store(emb_user + (long)user[r] * 128 + t * 16, xr + t * 16);
    cvt16_store(emb_item + (long)item[r] * 128 + t * 16, xr + 128 + t * 16);

    pool_row(emb_k, pk + r * KNN, vpk[r], xr + 256,       t);
    pool_row(emb_k, tk + r * KNN, vtk[r], xr + 256 + 64,  t);
    pool_row(emb_k, ik + r * KNN, vik[r], xr + 256 + 128, t);
  }
  __syncthreads();

  // ---------------- layer 1: [32,448] x [448,256] -> H1 ---------------------
  {
    const int n0 = wv * 32;            // this wave: N-tiles n0 and n0+16
    v8f c00 = bias_frag(b1, n0, lane);
    v8f c01 = bias_frag(b1, n0 + 16, lane);
    v8f c10 = c00, c11 = c01;
#pragma unroll 2
    for (int kc = 0; kc < IN0 / 32; ++kc) {
      v16h a0 = ld_a_lds(sX, XSTR, 0,  kc, lane);
      v16h a1 = ld_a_lds(sX, XSTR, 16, kc, lane);
      v16h bA = ld_b_glb(W1t, IN0, n0,      kc, lane);
      v16h bB = ld_b_glb(W1t, IN0, n0 + 16, kc, lane);
      c00 = WMMA_F16(a0, bA, c00);
      c01 = WMMA_F16(a0, bB, c01);
      c10 = WMMA_F16(a1, bA, c10);
      c11 = WMMA_F16(a1, bB, c11);
    }
    st_c_relu(sH1, H1STR, 0,  n0,      lane, c00);
    st_c_relu(sH1, H1STR, 0,  n0 + 16, lane, c01);
    st_c_relu(sH1, H1STR, 16, n0,      lane, c10);
    st_c_relu(sH1, H1STR, 16, n0 + 16, lane, c11);
  }
  __syncthreads();

  // ---------------- layer 2: [32,256] x [256,128] -> H2 ---------------------
  {
    const int n0 = wv * 16;
    v8f c0 = bias_frag(b2, n0, lane);
    v8f c1 = c0;
#pragma unroll 2
    for (int kc = 0; kc < H1DIM / 32; ++kc) {
      v16h a0 = ld_a_lds(sH1, H1STR, 0,  kc, lane);
      v16h a1 = ld_a_lds(sH1, H1STR, 16, kc, lane);
      v16h bA = ld_b_glb(W2t, H1DIM, n0, kc, lane);
      c0 = WMMA_F16(a0, bA, c0);
      c1 = WMMA_F16(a1, bA, c1);
    }
    st_c_relu(sH2, H2STR, 0,  n0, lane, c0);
    st_c_relu(sH2, H2STR, 16, n0, lane, c1);
  }
  __syncthreads();

  // ---------------- layer 3: [32,128] x [128,64] -> H3 ----------------------
  {
    const int m0 = (wv >> 2) * 16;
    const int n0 = (wv & 3) * 16;
    v8f c0 = bias_frag(b3, n0, lane);
#pragma unroll
    for (int kc = 0; kc < H2DIM / 32; ++kc) {
      v16h a0 = ld_a_lds(sH2, H2STR, m0, kc, lane);
      v16h bA = ld_b_glb(W3t, H2DIM, n0, kc, lane);
      c0 = WMMA_F16(a0, bA, c0);
    }
    st_c_relu(sH3, H3STR, m0, n0, lane, c0);
  }
  __syncthreads();

  // ---------------- layer 4: [32,64] . Wp + bp -> out -----------------------
  if (tid < TILE_B) {
    const _Float16* h = sH3 + tid * H3STR;
    float acc = bp[0];
#pragma unroll
    for (int c = 0; c < H3DIM / 8; ++c) {
      v8h hv = *(const v8h*)(h + c * 8);      // 16B-aligned ds_load_b128
      v4f w0 = *(const v4f*)(Wp + c * 8);
      v4f w1 = *(const v4f*)(Wp + c * 8 + 4);
#pragma unroll
      for (int j = 0; j < 4; ++j) {
        acc += (float)hv[j]     * w0[j];
        acc += (float)hv[4 + j] * w1[j];
      }
    }
    out[(long)blockIdx.x * TILE_B + tid] = acc;
  }
}

extern "C" void kernel_launch(void* const* d_in, const int* in_sizes, int n_in,
                              void* d_out, int out_size, void* d_ws, size_t ws_size,
                              hipStream_t stream) {
  const int*   user     = (const int*)d_in[0];
  const int*   item     = (const int*)d_in[1];
  const int*   pk       = (const int*)d_in[2];
  const int*   tk       = (const int*)d_in[3];
  const int*   ik       = (const int*)d_in[4];
  const int*   vpk      = (const int*)d_in[5];
  const int*   vtk      = (const int*)d_in[6];
  const int*   vik      = (const int*)d_in[7];
  const float* emb_user = (const float*)d_in[8];
  const float* emb_item = (const float*)d_in[9];
  const float* emb_k    = (const float*)d_in[10];
  const float* W1       = (const float*)d_in[11];
  const float* b1       = (const float*)d_in[12];
  const float* W2       = (const float*)d_in[13];
  const float* b2       = (const float*)d_in[14];
  const float* W3       = (const float*)d_in[15];
  const float* b3       = (const float*)d_in[16];
  const float* Wp       = (const float*)d_in[17];
  const float* bp       = (const float*)d_in[18];
  float*       out      = (float*)d_out;

  _Float16* wsH = (_Float16*)d_ws;

  kd_prep_weights<<<(WS_TOTAL + 255) / 256, 256, 0, stream>>>(W1, W2, W3, wsH);

  kd_fused<<<B_TOTAL / TILE_B, 256, 0, stream>>>(
      user, item, pk, tk, ik, vpk, vtk, vik,
      emb_user, emb_item, emb_k,
      wsH + W1T_OFF, b1, wsH + W2T_OFF, b2, wsH + W3T_OFF, b3,
      Wp, bp, out);
}
